// CliffordEngine_8186207666950
// MI455X (gfx1250) — matrix operand order repaired and checked by reading, compiled, not run on hardware
//
#include <hip/hip_runtime.h>
#include <hip/hip_bf16.h>
#include <stdint.h>

// ---------------------------------------------------------------------------
// CliffordEngine on MI455X (gfx1250, wave32, WMMA).
//  state [B=16,F=8,C=8,G=32,G=32,G=32] fp32, P=4 passes.
//  Per pass: grouped causal 3^3 conv (8x8 comps x 27 taps per field) -> tanh
//            -> 8x8 field mix -> alpha residual.  Final sigmoid gate.
//  Conv contraction mapped to v_wmma_f32_16x16x32_bf16:
//    C[pos(16), outcomp(16; 8 used)] += A[pos, K] x B[K, outcomp]
//    K = tap_local(4) * 8 + incomp(8); 7 WMMAs cover 27 taps (tap 27 zero-B).
//  A fragment per lane = two ds_load_b128 from an LDS tile stored
//  [field][z][y][x][comp] with 8 comps packed as 8 bf16 = 16B per voxel.
//  Two M-tiles (x-halves) processed concurrently: two independent WMMA
//  accumulation chains, tile-1 loads = tile-0 addresses + 256B immediate.
//  Transcendentals are branchless (v_tanh_f32 if available, else exp2/rcp).
//  Workspace requirement: 128MB ping buffer + ~256KB weights (~134.5 MB).
// ---------------------------------------------------------------------------

#define GDIM   32
#define G3     32768
#define FNUM   8
#define CNUM   8
#define PNUM   4
#define TZ     4
#define TY     4
#define ZH     6       // z tile + halo
#define YH     6
#define XH     34      // x extent incl. halo
#define XS     36      // padded LDS x stride (voxels)

#define XTILE_BYTES (FNUM*ZH*YH*XS*16)            // 165888 (bf16 voxels, 16B each)
#define YTILE_BYTES (FNUM*TZ*TY*GDIM*CNUM*2)      // 65536  (bf16 y buffer)
#define MIX_BYTES   (64*4)
#define SHMEM_BYTES (XTILE_BYTES + YTILE_BYTES + MIX_BYTES)   // 231680 <= 320KB WGP LDS

typedef __attribute__((ext_vector_type(16))) __bf16 v16bf;
typedef __attribute__((ext_vector_type(8)))  float  v8f;

union U32x8 { uint4 q[2]; v16bf v; };

// Cl(3,0) Cayley: W_eff[i,j,c] = c_sign[i][j] * w[c, c_kidx[i][j]]
__device__ __constant__ int c_kidx[64] = {
  0,1,2,3,4,5,6,7,
  1,0,4,5,2,3,7,6,
  2,4,0,6,1,7,3,5,
  3,5,6,0,7,1,2,4,
  4,2,1,7,0,6,5,3,
  5,3,7,1,6,0,4,2,
  6,7,3,2,5,4,0,1,
  7,6,5,4,3,2,1,0 };
__device__ __constant__ float c_sign[64] = {
  1, 1, 1, 1,-1,-1,-1,-1,
  1, 1, 1, 1,-1,-1,-1,-1,
  1,-1, 1, 1, 1, 1,-1, 1,
  1,-1,-1, 1,-1, 1, 1,-1,
  1,-1, 1, 1, 1, 1,-1, 1,
  1,-1,-1, 1,-1, 1, 1,-1,
  1, 1,-1, 1, 1,-1, 1, 1,
  1, 1,-1, 1, 1,-1, 1, 1 };

__device__ inline unsigned short f2bf(float x) {
  unsigned u = __float_as_uint(x);
  u += 0x7FFFu + ((u >> 16) & 1u);      // round-to-nearest-even
  return (unsigned short)(u >> 16);
}
__device__ inline float bf2f(unsigned short h) {
  return __uint_as_float(((unsigned)h) << 16);
}
__device__ inline unsigned pk2(float lo, float hi) {
  return (unsigned)f2bf(lo) | ((unsigned)f2bf(hi) << 16);
}

// Branchless transcendentals (HW TRANS ops; no libm exec-mask churn).
__device__ inline float tanh_fast(float x) {
#if __has_builtin(__builtin_amdgcn_tanhf)
  return __builtin_amdgcn_tanhf(x);            // v_tanh_f32 (gfx1250 TRANS)
#else
  // tanh(x) = 1 - 2/(exp2(2*log2e*x)+1); exact limits via rcp(inf)=0.
  float e = __builtin_amdgcn_exp2f(x * 2.8853900817779268f);
  return 1.f - 2.f * __builtin_amdgcn_rcpf(e + 1.f);
#endif
}
__device__ inline float sigmoid_fast(float x) {
  float e = __builtin_amdgcn_exp2f(x * -1.4426950408889634f);
  return __builtin_amdgcn_rcpf(1.f + e);
}

// ---------------------------------------------------------------------------
// Setup: expand all_weights [F,P,27,8] through the Cayley table directly into
// per-lane WMMA B-fragment layout: [(p,f,t,lane)] x 8 dwords (16 bf16).
// B is 32(K)x16(N): lane L holds column N=L%16, K half = (L<16 ? 0..15 : 16..31),
// VGPR v packs K = half+2v (lo), half+2v+1 (hi).
// ---------------------------------------------------------------------------
__global__ void setup_bfrag(const float* __restrict__ w, uint4* __restrict__ bfrag) {
  int id = blockIdx.x * 256 + threadIdx.x;
  if (id >= PNUM * FNUM * 7 * 32) return;
  int lane = id & 31;
  int t    = (id >> 5) % 7;
  int f    = (id / (32 * 7)) % FNUM;
  int p    =  id / (32 * 7 * FNUM);
  int i     = lane & 15;
  int khalf = (lane < 16) ? 0 : 16;
  unsigned d[8];
#pragma unroll
  for (int v = 0; v < 8; v++) {
    float vals[2];
#pragma unroll
    for (int e = 0; e < 2; e++) {
      int K  = khalf + 2 * v + e;
      int cl = K >> 3;
      int j  = K & 7;
      int c  = 4 * t + cl;
      float val = 0.f;
      if (i < 8 && c < 27) {
        int kk = c_kidx[i * 8 + j];
        val = c_sign[i * 8 + j] * w[((size_t)(f * PNUM + p) * 27 + c) * 8 + kk];
      }
      vals[e] = val;
    }
    d[v] = pk2(vals[0], vals[1]);
  }
  uint4* dst = bfrag + (size_t)id * 2;
  dst[0] = make_uint4(d[0], d[1], d[2], d[3]);
  dst[1] = make_uint4(d[4], d[5], d[6], d[7]);
}

// softmax of field_mix_logits over last axis + sigmoid(alpha logits)
__global__ void setup_mix(const float* __restrict__ logits,
                          const float* __restrict__ alog,
                          float* __restrict__ mixTab, float* __restrict__ alphaTab) {
  int id = threadIdx.x;
  if (id < 32) {
    int p = id >> 3, g = id & 7;
    const float* row = logits + (size_t)(p * 8 + g) * 8;
    float m = row[0];
    for (int f = 1; f < 8; f++) m = fmaxf(m, row[f]);
    float e[8], s = 0.f;
    for (int f = 0; f < 8; f++) {
      e[f] = __builtin_amdgcn_exp2f((row[f] - m) * 1.4426950408889634f);
      s += e[f];
    }
    for (int f = 0; f < 8; f++) mixTab[p * 64 + g * 8 + f] = e[f] / s;
  } else if (id < 32 + PNUM) {
    int p = id - 32;
    alphaTab[p] = sigmoid_fast(alog[p]);
  }
}

// ---------------------------------------------------------------------------
// One pass: conv(WMMA) + tanh -> LDS, then field-mix + residual -> x_out.
// Grid: 1024 blocks (16 batch x 8 z-tiles x 8 y-tiles), 512 threads (16 waves).
// ---------------------------------------------------------------------------
__global__ __launch_bounds__(512) void conv_mix_pass(
    const float* __restrict__ x_in, const float* __restrict__ biases,
    const uint4* __restrict__ bfrag, const float* __restrict__ mixTab,
    const float* __restrict__ alphaTab, float* __restrict__ x_out, int pass) {
  extern __shared__ __align__(16) char smem[];
  char*           xtile = smem;                                   // bf16 voxels
  unsigned short* ytile = (unsigned short*)(smem + XTILE_BYTES);  // tanh'ed y
  float*          mixL  = (float*)(smem + XTILE_BYTES + YTILE_BYTES);

  const int tid = threadIdx.x;
  const int bid = blockIdx.x;
  const int bb  = bid >> 6;
  const int rem = bid & 63;
  const int z0  = (rem >> 3) * TZ;
  const int y0  = (rem & 7) * TY;

  if (tid < 64) mixL[tid] = mixTab[pass * 64 + tid];
  const float alpha = alphaTab[pass];

  // ---- stage 1: load haloed tile (all fields), transpose comps to inner, bf16
  const int nvox = FNUM * ZH * YH * XH;   // 9792
  for (int n = tid; n < nvox; n += 512) {
    int f  = n / (ZH * YH * XH);
    int r  = n - f * (ZH * YH * XH);
    int lz = r / (YH * XH);  r -= lz * (YH * XH);
    int ly = r / XH;
    int lx = r - ly * XH;
    int gz = z0 - 2 + lz, gy = y0 - 2 + ly, gx = lx - 2;
    uint4 pack = make_uint4(0u, 0u, 0u, 0u);
    if (gz >= 0 && gy >= 0 && gx >= 0) {
      const float* src = x_in + ((size_t)(bb * FNUM + f) * CNUM) * G3
                              + (size_t)gz * 1024 + (size_t)gy * 32 + gx;
      float v[8];
#pragma unroll
      for (int j = 0; j < 8; j++) v[j] = src[(size_t)j * G3];
      pack = make_uint4(pk2(v[0], v[1]), pk2(v[2], v[3]), pk2(v[4], v[5]), pk2(v[6], v[7]));
    }
    *(uint4*)(xtile + ((size_t)((f * ZH + lz) * YH + ly) * XS + lx) * 16) = pack;
  }
  __syncthreads();

  // ---- stage 2: per-field conv via dual interleaved WMMA chains
  const int wave    = tid >> 5;
  const int lane    = tid & 31;
  const int f       = wave >> 1;        // 2 waves per field
  const int mbase   = (wave & 1) * 16;  // 16 M-tiles each (8 pairs)
  const int i       = lane & 15;        // out-comp column (valid < 8)
  const int laneLow = (lane < 16) ? 1 : 0;

  v16bf Bf[7];
#pragma unroll
  for (int t = 0; t < 7; t++) {
    const uint4* bp = bfrag + ((size_t)((pass * FNUM + f) * 7 + t) * 32 + lane) * 2;
    U32x8 u; u.q[0] = bp[0]; u.q[1] = bp[1];
    Bf[t] = u.v;
  }
  const float bias = (i < 8) ? biases[(size_t)(f * PNUM + pass) * 8 + i] : 0.f;

  // per-lane tap byte-offsets (lane-half selects K-subset taps), hoisted
  int offT[7][2];
#pragma unroll
  for (int t = 0; t < 7; t++) {
    int ca = t * 4 + (laneLow ? 0 : 2);
    int cb = t * 4 + (laneLow ? 1 : 3);
    if (cb > 26) cb = 26;               // padded tap: B is zero there
    offT[t][0] = (((ca / 9) * YH + (ca / 3) % 3) * XS + ca % 3) * 16;
    offT[t][1] = (((cb / 9) * YH + (cb / 3) % 3) * XS + cb % 3) * 16;
  }

  for (int jj = 0; jj < 16; jj += 2) {  // pair of x-half tiles: +0 and +256B
    int m  = mbase + jj;
    int oz = m >> 3;
    int oy = (m >> 1) & 3;
    const char* base = xtile + ((size_t)((f * ZH + oz) * YH + oy) * XS + i) * 16;

    v8f acc0 = {}, acc1 = {};
#pragma unroll
    for (int t = 0; t < 7; t++) {
      const char* pA = base + offT[t][0];
      const char* pB = base + offT[t][1];
      U32x8 a0, a1;
      a0.q[0] = *(const uint4*)(pA);
      a0.q[1] = *(const uint4*)(pB);
      a1.q[0] = *(const uint4*)(pA + 256);   // x0 + 16 voxels
      a1.q[1] = *(const uint4*)(pB + 256);
      acc0 = __builtin_amdgcn_wmma_f32_16x16x32_bf16(
          false, a0.v, false, Bf[t], (short)0, acc0, false, false);
      acc1 = __builtin_amdgcn_wmma_f32_16x16x32_bf16(
          false, a1.v, false, Bf[t], (short)0, acc1, false, false);
    }
    if (i < 8) {
      unsigned short* yrow = ytile + (size_t)(((f * TZ + oz) * TY + oy) * GDIM) * CNUM + i;
#pragma unroll
      for (int r = 0; r < 8; r++) {
        int M = r + (laneLow ? 0 : 8);        // C row -> position within half
        yrow[(size_t)M * CNUM]        = f2bf(tanh_fast(acc0[r] + bias));
        yrow[(size_t)(16 + M) * CNUM] = f2bf(tanh_fast(acc1[r] + bias));
      }
    }
  }
  __syncthreads();

  // ---- stage 3: 8x8 field mix + alpha residual, coalesced global I/O
  for (int q = tid; q < 4096; q += 512) {   // (comp, voxel) pairs
    int comp = q >> 9;
    int v    = q & 511;
    int z = v >> 7, yy = (v >> 5) & 3, x = v & 31;
    float yv[8];
#pragma unroll
    for (int ff = 0; ff < 8; ff++)
      yv[ff] = bf2f(ytile[(size_t)(((ff * TZ + z) * TY + yy) * GDIM + x) * CNUM + comp]);
    size_t sidx = (size_t)(z0 + z) * 1024 + (size_t)(y0 + yy) * 32 + x;
#pragma unroll
    for (int g = 0; g < 8; g++) {
      float xm = 0.f;
#pragma unroll
      for (int ff = 0; ff < 8; ff++) xm = fmaf(mixL[g * 8 + ff], yv[ff], xm);
      size_t gi = ((size_t)(bb * FNUM + g) * CNUM + comp) * G3 + sidx;
      float xb = x_in[gi];
      x_out[gi] = alpha * xb + (1.f - alpha) * xm;
    }
  }
}

// ---------------------------------------------------------------------------
// Final gate (pointwise, in-place on x buffer), float4-vectorized.
// f/c indices are uniform within a quad (G3 multiple of 4).
// ---------------------------------------------------------------------------
__global__ __launch_bounds__(256) void gate_kernel(
    const float4* __restrict__ old_s, float4* __restrict__ xio,
    const float* __restrict__ gw, const float* __restrict__ gb, size_t n4) {
  size_t q = (size_t)blockIdx.x * 256 + threadIdx.x;
  if (q >= n4) return;
  size_t idx = q * 4;
  int c = (int)((idx >> 15) & 7);
  int f = (int)((idx >> 18) & 7);
  float w = gw[f * 8 + c], b = gb[f * 8 + c];
  float4 o = old_s[q];
  float4 x = xio[q];
  float g0 = sigmoid_fast(w * o.x + b);
  float g1 = sigmoid_fast(w * o.y + b);
  float g2 = sigmoid_fast(w * o.z + b);
  float g3 = sigmoid_fast(w * o.w + b);
  x.x = g0 * o.x + (1.f - g0) * x.x;
  x.y = g1 * o.y + (1.f - g1) * x.y;
  x.z = g2 * o.z + (1.f - g2) * x.z;
  x.w = g3 * o.w + (1.f - g3) * x.w;
  xio[q] = x;
}

// ---------------------------------------------------------------------------
extern "C" void kernel_launch(void* const* d_in, const int* in_sizes, int n_in,
                              void* d_out, int out_size, void* d_ws, size_t ws_size,
                              hipStream_t stream) {
  const float* state   = (const float*)d_in[0];
  const float* weights = (const float*)d_in[1];
  const float* biases  = (const float*)d_in[2];
  const float* mixlog  = (const float*)d_in[3];
  const float* alog    = (const float*)d_in[4];
  const float* gw      = (const float*)d_in[5];
  const float* gb      = (const float*)d_in[6];
  float* out = (float*)d_out;

  const size_t n = (size_t)out_size;             // 33,554,432 elements
  const size_t stateBytes = n * sizeof(float);   // 128 MB

  // workspace: [ping x buffer | B fragments | mix table | alphas]  (~134.5 MB)
  char*  ws       = (char*)d_ws;
  float* xping    = (float*)ws;
  uint4* bfrag    = (uint4*)(ws + stateBytes);
  const size_t bfragBytes = (size_t)PNUM * FNUM * 7 * 32 * 32;   // 229,376 B
  float* mixTab   = (float*)(ws + stateBytes + bfragBytes);
  float* alphaTab = mixTab + PNUM * 64;

  // weight expansion + mix/alpha precompute
  setup_bfrag<<<(PNUM * FNUM * 7 * 32 + 255) / 256, 256, 0, stream>>>(weights, bfrag);
  setup_mix<<<1, 64, 0, stream>>>(mixlog, alog, mixTab, alphaTab);

  (void)hipFuncSetAttribute(reinterpret_cast<const void*>(conv_mix_pass),
                            hipFuncAttributeMaxDynamicSharedMemorySize,
                            (int)SHMEM_BYTES);

  // 4 passes, ping-pong: state -> ws -> out -> ws -> out
  const float* src[PNUM] = { state, xping, out, xping };
  float*       dst[PNUM] = { xping, out, xping, out };
  for (int p = 0; p < PNUM; p++) {
    conv_mix_pass<<<1024, 512, SHMEM_BYTES, stream>>>(
        src[p], biases, bfrag, mixTab, alphaTab, dst[p], p);
  }

  size_t n4 = n / 4;
  gate_kernel<<<(unsigned)((n4 + 255) / 256), 256, 0, stream>>>(
      (const float4*)state, (float4*)out, gw, gb, n4);
}
